// ConformerRNNT_69337952026795
// MI455X (gfx1250) — matrix operand order repaired and checked by reading, not compiled, not run on hardware
//
#include <hip/hip_runtime.h>
#include <hip/hip_bf16.h>
#include <math.h>

#define B_     4
#define T_     256
#define U_     64
#define IN_DIM 80
#define D_     256
#define FFN_   1024
#define KW_    31
#define H_     4
#define L_     4
#define HD_    64
#define PADW_  15
#define VOCAB_ 500
#define DEC_   256
#define JOINT_ 512
#define BT     (B_*T_)
#define BU     (B_*U_)

typedef __attribute__((ext_vector_type(16))) __bf16 v16bf;
typedef __attribute__((ext_vector_type(8)))  float  v8f;

enum { EPI_NONE = 0, EPI_SILU = 1, EPI_RES = 2 };

__device__ __forceinline__ float sigmf(float x) { return 1.f / (1.f + __expf(-x)); }

union BFRAG { v16bf v; uint4 q[2]; };

__device__ __forceinline__ void cvtrelu4(v16bf& v, int i, float4 e, float4 d) {
  v[i]   = (__bf16)fmaxf(e.x + d.x, 0.f);
  v[i+1] = (__bf16)fmaxf(e.y + d.y, 0.f);
  v[i+2] = (__bf16)fmaxf(e.z + d.z, 0.f);
  v[i+3] = (__bf16)fmaxf(e.w + d.w, 0.f);
}
#define LD4(p)  (*(const float4*)(p))
#define LDU4(p) (*(const uint4*)(p))

// ---------------------------------------------------------------------------
// f32 -> bf16 tensor convert (weights pre-pass, enc convert)
// ---------------------------------------------------------------------------
__global__ void cvtw_k(const float* __restrict__ in, __bf16* __restrict__ out, int n) {
  int i = blockIdx.x * blockDim.x + threadIdx.x;
  if (i < n) out[i] = (__bf16)in[i];
}

// ---------------------------------------------------------------------------
// K-slice register set for the GEMM ping-pong pipeline (no copies).
// ---------------------------------------------------------------------------
struct KSet { uint4 a0, a1, b0[4], b1[4]; };

__device__ __forceinline__ void load_set(const __bf16* __restrict__ arow,
                                         const __bf16* const* wrow,
                                         int kk, int half, KSet& s) {
  const __bf16* ap = arow + kk + half * 8;
  s.a0 = LDU4(ap); s.a1 = LDU4(ap + 16);
#pragma unroll
  for (int j = 0; j < 4; ++j) {
    const __bf16* wp = wrow[j] + kk + half * 8;
    s.b0[j] = LDU4(wp); s.b1[j] = LDU4(wp + 16);
  }
}

__device__ __forceinline__ void wmma_set(const KSet& s, v8f& c0, v8f& c1, v8f& c2, v8f& c3) {
  BFRAG av; av.q[0] = s.a0; av.q[1] = s.a1;
  BFRAG bv;
  bv.q[0] = s.b0[0]; bv.q[1] = s.b1[0];
  c0 = __builtin_amdgcn_wmma_f32_16x16x32_bf16(false, av.v, false, bv.v, (short)0, c0, false, false);
  bv.q[0] = s.b0[1]; bv.q[1] = s.b1[1];
  c1 = __builtin_amdgcn_wmma_f32_16x16x32_bf16(false, av.v, false, bv.v, (short)0, c1, false, false);
  bv.q[0] = s.b0[2]; bv.q[1] = s.b1[2];
  c2 = __builtin_amdgcn_wmma_f32_16x16x32_bf16(false, av.v, false, bv.v, (short)0, c2, false, false);
  bv.q[0] = s.b0[3]; bv.q[1] = s.b1[3];
  c3 = __builtin_amdgcn_wmma_f32_16x16x32_bf16(false, av.v, false, bv.v, (short)0, c3, false, false);
}

// ---------------------------------------------------------------------------
// GEMM: out = epi( A[M,Kd](bf16) @ W[N,Kd]^T(bf16) + bias ), f32 accumulate.
// Wave: 16 rows x 64 cols (4 accs). Block: 32 rows x 256 cols. Kd % 64 == 0.
// Manual 2x K-unroll with ping-pong register sets: loads feed WMMA directly.
// ---------------------------------------------------------------------------
__global__ __launch_bounds__(256)
void gemm_wmma_bf16(const __bf16* __restrict__ A, int lda,
                    const __bf16* __restrict__ W, int ldw,
                    const float* __restrict__ bias,
                    const float* __restrict__ res,
                    float* __restrict__ outF, __bf16* __restrict__ outB,
                    int ldo, int M, int N, int Kd, int epi, float alpha)
{
  const int lane = threadIdx.x & 31;
  const int wave = threadIdx.x >> 5;
  const int half = lane >> 4;
  const int l16  = lane & 15;
  const int rowBase = blockIdx.x * 32 + (wave >> 2) * 16;
  const int colBase = blockIdx.y * 256 + (wave & 3) * 64;

  int m = rowBase + l16; if (m >= M) m = M - 1;
  const __bf16* __restrict__ arow = A + (size_t)m * lda;

  int ncol[4];
  const __bf16* wrow[4];
#pragma unroll
  for (int j = 0; j < 4; ++j) {
    ncol[j] = colBase + j * 16 + l16;
    int nc = ncol[j] < N ? ncol[j] : N - 1;   // clamp; OOB lanes never stored
    wrow[j] = W + (size_t)nc * ldw;
  }

  KSet s0, s1;
  load_set(arow, wrow, 0, half, s0);
  v8f acc0 = {}, acc1 = {}, acc2 = {}, acc3 = {};

  for (int k = 0; k < Kd; k += 64) {
    load_set(arow, wrow, k + 32, half, s1);        // prefetch slice k+32
    wmma_set(s0, acc0, acc1, acc2, acc3);          // consume slice k
    if (k + 64 < Kd)
      load_set(arow, wrow, k + 64, half, s0);      // prefetch slice k+64
    wmma_set(s1, acc0, acc1, acc2, acc3);          // consume slice k+32
  }

  v8f accs[4] = {acc0, acc1, acc2, acc3};
#pragma unroll
  for (int j = 0; j < 4; ++j) {
    const int col = ncol[j];
    if (col >= N) continue;
    float bb = bias ? bias[col] : 0.f;
#pragma unroll
    for (int i = 0; i < 8; ++i) {
      int row = rowBase + half * 8 + i;
      if (row < M) {
        float s = accs[j][i] + bb;
        if (epi == EPI_SILU)     s = s * sigmf(s);
        else if (epi == EPI_RES) s = res[(size_t)row * ldo + col] + alpha * s;
        if (outF) outF[(size_t)row * ldo + col] = s;
        if (outB) outB[(size_t)row * ldo + col] = (__bf16)s;
      }
    }
  }
}

// ---------------------------------------------------------------------------
// Joint logits: out[b,t,u,v] = relu(e'[b,t,:]+d[b,u,:]) @ jw2^T + jb2
// where e' already includes jb1 (folded into e-projection bias).
// A fused on the fly (f32 add+relu -> cvt_pk); B bf16, ping-pong prefetched.
// Wave: 16 rows x 64 cols; block: 32 rows x 256 cols; grid (2048, 2).
// ---------------------------------------------------------------------------
struct BSet { uint4 b0[4], b1[4]; };

__device__ __forceinline__ void load_bset(const __bf16* const* wrow, int kk, int half, BSet& s) {
#pragma unroll
  for (int j = 0; j < 4; ++j) {
    const __bf16* wp = wrow[j] + kk + half * 8;
    s.b0[j] = LDU4(wp); s.b1[j] = LDU4(wp + 16);
  }
}

__device__ __forceinline__ void wmma_bset(const BFRAG& av, const BSet& s,
                                          v8f& c0, v8f& c1, v8f& c2, v8f& c3) {
  BFRAG bv;
  bv.q[0] = s.b0[0]; bv.q[1] = s.b1[0];
  c0 = __builtin_amdgcn_wmma_f32_16x16x32_bf16(false, av.v, false, bv.v, (short)0, c0, false, false);
  bv.q[0] = s.b0[1]; bv.q[1] = s.b1[1];
  c1 = __builtin_amdgcn_wmma_f32_16x16x32_bf16(false, av.v, false, bv.v, (short)0, c1, false, false);
  bv.q[0] = s.b0[2]; bv.q[1] = s.b1[2];
  c2 = __builtin_amdgcn_wmma_f32_16x16x32_bf16(false, av.v, false, bv.v, (short)0, c2, false, false);
  bv.q[0] = s.b0[3]; bv.q[1] = s.b1[3];
  c3 = __builtin_amdgcn_wmma_f32_16x16x32_bf16(false, av.v, false, bv.v, (short)0, c3, false, false);
}

__global__ __launch_bounds__(256)
void joint_wmma(const float* __restrict__ e, const float* __restrict__ dj,
                const __bf16* __restrict__ jw2b, const float* __restrict__ jb2,
                float* __restrict__ out)
{
  const int lane = threadIdx.x & 31;
  const int wave = threadIdx.x >> 5;
  const int half = lane >> 4;
  const int l16  = lane & 15;
  const int rowBase = blockIdx.x * 32 + (wave >> 2) * 16;
  const int colBase = blockIdx.y * 256 + (wave & 3) * 64;

  const int m = rowBase + l16;            // row in [0, B*T*U)
  const int b = m >> 14;                  // / (T*U) = 16384
  const int t = (m >> 6) & (T_ - 1);
  const int u = m & (U_ - 1);
  const float* __restrict__ erow = e  + (size_t)(b * T_ + t) * JOINT_;
  const float* __restrict__ drow = dj + (size_t)(b * U_ + u) * JOINT_;

  int ncol[4];
  const __bf16* wrow[4];
#pragma unroll
  for (int j = 0; j < 4; ++j) {
    ncol[j] = colBase + j * 16 + l16;
    int nc = ncol[j] < VOCAB_ ? ncol[j] : VOCAB_ - 1;
    wrow[j] = jw2b + (size_t)nc * JOINT_;
  }

  BSet B0, B1;
  load_bset(wrow, 0, half, B0);
  v8f acc0 = {}, acc1 = {}, acc2 = {}, acc3 = {};

  for (int k = 0; k < JOINT_; k += 64) {
    load_bset(wrow, k + 32, half, B1);
    // issue raw A loads for BOTH half-slices before any conversion
    const int ka0 = k + half * 8;
    const int ka1 = k + 32 + half * 8;
    float4 e00 = LD4(erow + ka0),      e01 = LD4(erow + ka0 + 4);
    float4 e02 = LD4(erow + ka0 + 16), e03 = LD4(erow + ka0 + 20);
    float4 d00 = LD4(drow + ka0),      d01 = LD4(drow + ka0 + 4);
    float4 d02 = LD4(drow + ka0 + 16), d03 = LD4(drow + ka0 + 20);
    float4 e10 = LD4(erow + ka1),      e11 = LD4(erow + ka1 + 4);
    float4 e12 = LD4(erow + ka1 + 16), e13 = LD4(erow + ka1 + 20);
    float4 d10 = LD4(drow + ka1),      d11 = LD4(drow + ka1 + 4);
    float4 d12 = LD4(drow + ka1 + 16), d13 = LD4(drow + ka1 + 20);

    BFRAG av0;
    cvtrelu4(av0.v, 0, e00, d00); cvtrelu4(av0.v, 4,  e01, d01);
    cvtrelu4(av0.v, 8, e02, d02); cvtrelu4(av0.v, 12, e03, d03);
    wmma_bset(av0, B0, acc0, acc1, acc2, acc3);

    if (k + 64 < JOINT_) load_bset(wrow, k + 64, half, B0);

    BFRAG av1;
    cvtrelu4(av1.v, 0, e10, d10); cvtrelu4(av1.v, 4,  e11, d11);
    cvtrelu4(av1.v, 8, e12, d12); cvtrelu4(av1.v, 12, e13, d13);
    wmma_bset(av1, B1, acc0, acc1, acc2, acc3);
  }

  v8f accs[4] = {acc0, acc1, acc2, acc3};
#pragma unroll
  for (int j = 0; j < 4; ++j) {
    const int col = ncol[j];
    if (col >= VOCAB_) continue;
    const float bb = jb2[col];
#pragma unroll
    for (int i = 0; i < 8; ++i) {
      int row = rowBase + half * 8 + i;
      out[(size_t)row * VOCAB_ + col] = accs[j][i] + bb;
    }
  }
}

// ---------------------------------------------------------------------------
// LayerNorm (D_=256), one row per block. f32-out and bf16-out variants.
// ---------------------------------------------------------------------------
__global__ __launch_bounds__(256)
void layernorm_k(const float* __restrict__ in, const float* __restrict__ g,
                 const float* __restrict__ b, float* __restrict__ out)
{
  __shared__ float red[256];
  const int row = blockIdx.x, tid = threadIdx.x;
  float x = in[(size_t)row * D_ + tid];
  red[tid] = x; __syncthreads();
  for (int off = 128; off > 0; off >>= 1) {
    if (tid < off) red[tid] += red[tid + off];
    __syncthreads();
  }
  float mean = red[0] * (1.f / D_); __syncthreads();
  float d = x - mean;
  red[tid] = d * d; __syncthreads();
  for (int off = 128; off > 0; off >>= 1) {
    if (tid < off) red[tid] += red[tid + off];
    __syncthreads();
  }
  float r = rsqrtf(red[0] * (1.f / D_) + 1e-5f);
  out[(size_t)row * D_ + tid] = d * r * g[tid] + b[tid];
}

__global__ __launch_bounds__(256)
void layernorm_bf16_k(const float* __restrict__ in, const float* __restrict__ g,
                      const float* __restrict__ b, __bf16* __restrict__ out)
{
  __shared__ float red[256];
  const int row = blockIdx.x, tid = threadIdx.x;
  float x = in[(size_t)row * D_ + tid];
  red[tid] = x; __syncthreads();
  for (int off = 128; off > 0; off >>= 1) {
    if (tid < off) red[tid] += red[tid + off];
    __syncthreads();
  }
  float mean = red[0] * (1.f / D_); __syncthreads();
  float d = x - mean;
  red[tid] = d * d; __syncthreads();
  for (int off = 128; off > 0; off >>= 1) {
    if (tid < off) red[tid] += red[tid + off];
    __syncthreads();
  }
  float r = rsqrtf(red[0] * (1.f / D_) + 1e-5f);
  out[(size_t)row * D_ + tid] = (__bf16)(d * r * g[tid] + b[tid]);
}

// ---------------------------------------------------------------------------
// Subsample conv1d: x[B,T,80] -> bf16 [B,T,256] (feeds lin GEMM only).
// ---------------------------------------------------------------------------
__global__ __launch_bounds__(256)
void subsample_k(const float* __restrict__ x, const float* __restrict__ w,
                 const float* __restrict__ bias, __bf16* __restrict__ out)
{
  __shared__ float xs[KW_ * IN_DIM];   // [k][c]
  const int bt = blockIdx.x;
  const int b = bt / T_, t = bt % T_;
  for (int i = threadIdx.x; i < KW_ * IN_DIM; i += 256) {
    int k = i / IN_DIM, c = i % IN_DIM;
    int tt = t + k - PADW_;
    xs[i] = (tt >= 0 && tt < T_) ? x[((size_t)(b * T_ + tt)) * IN_DIM + c] : 0.f;
  }
  __syncthreads();
  const int d = threadIdx.x;
  const float* wr = w + (size_t)d * IN_DIM * KW_;   // w[d,c,k]
  float acc = bias[d];
  for (int c = 0; c < IN_DIM; ++c) {
#pragma unroll
    for (int k = 0; k < KW_; ++k)
      acc += xs[k * IN_DIM + c] * wr[c * KW_ + k];
  }
  out[(size_t)bt * D_ + d] = (__bf16)acc;
}

// ---------------------------------------------------------------------------
// Attention: one block per (q, b*H+h). qkv f32; output bf16 (feeds out-proj).
// ---------------------------------------------------------------------------
__global__ __launch_bounds__(256)
void attention_k(const float* __restrict__ qkv, const int* __restrict__ xlen,
                 __bf16* __restrict__ out)
{
  __shared__ float qv[HD_];
  __shared__ float sc[T_];
  __shared__ float red[256];
  const int q = blockIdx.x;
  const int b = blockIdx.y / H_, h = blockIdx.y % H_;
  const int tid = threadIdx.x;

  if (tid < HD_) qv[tid] = qkv[((size_t)(b * T_ + q)) * (3 * D_) + h * HD_ + tid];
  __syncthreads();

  const float* krow = qkv + ((size_t)(b * T_ + tid)) * (3 * D_) + D_ + h * HD_;
  float s = 0.f;
#pragma unroll 8
  for (int d = 0; d < HD_; ++d) s += qv[d] * krow[d];
  s *= 0.125f;                                  // 1/sqrt(64)
  if (tid >= xlen[b]) s += -1e9f;               // key padding bias

  red[tid] = s; __syncthreads();
  for (int off = 128; off > 0; off >>= 1) {
    if (tid < off) red[tid] = fmaxf(red[tid], red[tid + off]);
    __syncthreads();
  }
  float mx = red[0]; __syncthreads();
  float p = __expf(s - mx);
  sc[tid] = p;
  red[tid] = p; __syncthreads();
  for (int off = 128; off > 0; off >>= 1) {
    if (tid < off) red[tid] += red[tid + off];
    __syncthreads();
  }
  float inv = 1.f / red[0];

  if (tid < HD_) {
    float o = 0.f;
    for (int k = 0; k < T_; ++k)
      o += sc[k] * qkv[((size_t)(b * T_ + k)) * (3 * D_) + 2 * D_ + h * HD_ + tid];
    out[((size_t)(b * T_ + q)) * D_ + h * HD_ + tid] = (__bf16)(o * inv);
  }
}

// ---------------------------------------------------------------------------
// GLU (f32 -> f32, feeds dwconv), dwconv+BN+SiLU (f32 -> bf16), embedding
// ---------------------------------------------------------------------------
__global__ void glu_k(const float* __restrict__ in, float* __restrict__ out) {
  int idx = blockIdx.x * blockDim.x + threadIdx.x;
  if (idx >= BT * D_) return;
  int r = idx / D_, d = idx % D_;
  float a = in[(size_t)r * (2 * D_) + d];
  float g = in[(size_t)r * (2 * D_) + D_ + d];
  out[idx] = a * sigmf(g);
}

__global__ __launch_bounds__(256)
void dwconv_k(const float* __restrict__ in, const float* __restrict__ w,
              const float* __restrict__ bias, const float* __restrict__ bng,
              const float* __restrict__ bnb, __bf16* __restrict__ out)
{
  const int bt = blockIdx.x, d = threadIdx.x;
  const int b = bt / T_, t = bt % T_;
  float acc = bias[d];
#pragma unroll
  for (int k = 0; k < KW_; ++k) {
    int tt = t + k - PADW_;
    if (tt >= 0 && tt < T_)
      acc += in[((size_t)(b * T_ + tt)) * D_ + d] * w[d * KW_ + k];
  }
  acc = acc * 0.9999950000374997f /* rsqrt(1+1e-5) */ * bng[d] + bnb[d];
  acc = acc * sigmf(acc);
  out[(size_t)bt * D_ + d] = (__bf16)acc;
}

__global__ void embed_k(const int* __restrict__ y, const float* __restrict__ emb,
                        float* __restrict__ out) {
  int idx = blockIdx.x * blockDim.x + threadIdx.x;
  if (idx >= BU * DEC_) return;
  int r = idx / DEC_, d = idx % DEC_;
  out[idx] = emb[(size_t)y[r] * DEC_ + d];
}

// ---------------------------------------------------------------------------
// Single-layer LSTM over U=64 steps, persistent single block (1024 threads).
// h/c state f32 in LDS; outputs bf16 (feed joint d-GEMM only).
// ---------------------------------------------------------------------------
__global__ __launch_bounds__(1024)
void lstm_k(const float* __restrict__ xin,  // [B,U,DEC]
            const float* __restrict__ wih, const float* __restrict__ whh,
            const float* __restrict__ bih, const float* __restrict__ bhh,
            __bf16* __restrict__ dec)       // [B,U,DEC]
{
  __shared__ float hs[B_][DEC_];
  __shared__ float cs[B_][DEC_];
  __shared__ float gs[B_][4 * DEC_];
  const int tid = threadIdx.x;              // 0..1023 -> gate index j
  if (tid < DEC_)
    for (int b = 0; b < B_; ++b) { hs[b][tid] = 0.f; cs[b][tid] = 0.f; }
  __syncthreads();

  const float* wi = wih + (size_t)tid * DEC_;
  const float* wh = whh + (size_t)tid * DEC_;

  for (int u = 0; u < U_; ++u) {
    for (int b = 0; b < B_; ++b) {
      const float* xv = xin + ((size_t)(b * U_ + u)) * DEC_;
      float g = bih[tid] + bhh[tid];
#pragma unroll 4
      for (int d = 0; d < DEC_; ++d) g += xv[d] * wi[d] + hs[b][d] * wh[d];
      gs[b][tid] = g;
    }
    __syncthreads();
    if (tid < DEC_) {
      for (int b = 0; b < B_; ++b) {
        float i = gs[b][tid], f = gs[b][DEC_ + tid];
        float g = gs[b][2 * DEC_ + tid], o = gs[b][3 * DEC_ + tid];
        float c = sigmf(f) * cs[b][tid] + sigmf(i) * tanhf(g);
        float h = sigmf(o) * tanhf(c);
        cs[b][tid] = c; hs[b][tid] = h;
        dec[((size_t)(b * U_ + u)) * DEC_ + tid] = (__bf16)h;
      }
    }
    __syncthreads();
  }
}

// ---------------------------------------------------------------------------
// Host driver
// ---------------------------------------------------------------------------
extern "C" void kernel_launch(void* const* d_in, const int* in_sizes, int n_in,
                              void* d_out, int out_size, void* d_ws, size_t ws_size,
                              hipStream_t stream)
{
  (void)in_sizes; (void)out_size; (void)ws_size;

  const float* x    = (const float*)d_in[0];
  const int*   xlen = (const int*)d_in[1];
  const int*   y    = (const int*)d_in[2];

  const float* P[140];
  for (int i = 0; i + 3 < n_in && i < 140; ++i) P[i] = (const float*)d_in[3 + i];

  const float* sub_w = P[0]; const float* sub_b = P[1];
  const float* lin_w = P[2]; const float* lin_b = P[3];
  const float* emb   = P[124];
  const float* w_ih  = P[125]; const float* w_hh = P[126];
  const float* b_ih  = P[127]; const float* b_hh = P[128];
  const float* jw1   = P[129]; const float* jb1  = P[130];
  const float* jw2   = P[131]; const float* jb2  = P[132];

  // ---- f32 workspace ----
  float* fws  = (float*)d_ws;
  float* X    = fws;                        // [BT, D]  residual stream
  float* T2   = X    + (size_t)BT * D_;     // [BT, FFN] f32 scratch (qkv / pw1)
  float* T3   = T2   + (size_t)BT * FFN_;   // [BT, D]  GLU out (f32, dwconv in)
  float* Eb   = T3   + (size_t)BT * D_;     // [BT, JOINT]  (includes jb1)
  float* Db   = Eb   + (size_t)BT * JOINT_; // [BU, JOINT]
  float* EMBb = Db   + (size_t)BU * JOINT_; // [BU, DEC]

  // ---- bf16 workspace (activations + pre-converted weights) ----
  __bf16* bws   = (__bf16*)(EMBb + (size_t)BU * DEC_);
  __bf16* T1b   = bws; bws += (size_t)BT * D_;    // LN / subsample out
  __bf16* T2b   = bws; bws += (size_t)BT * FFN_;  // FFN hidden (SiLU, bf16)
  __bf16* T3b   = bws; bws += (size_t)BT * D_;    // attn out / dwconv out
  __bf16* Xb    = bws; bws += (size_t)BT * D_;    // final enc, bf16
  __bf16* DEC16 = bws; bws += (size_t)BU * DEC_;  // LSTM out, bf16
  __bf16* linb  = bws; bws += (size_t)D_ * D_;
  static const int wIdx[8] = {2, 4, 8, 10, 14, 20, 24, 26};
  static const int wSz [8] = {FFN_*D_, D_*FFN_, 3*D_*D_, D_*D_, 2*D_*D_, D_*D_, FFN_*D_, D_*FFN_};
  __bf16* Wb[L_][8];
  for (int l = 0; l < L_; ++l)
    for (int wq = 0; wq < 8; ++wq) { Wb[l][wq] = bws; bws += wSz[wq]; }
  __bf16* jw1b = bws; bws += (size_t)JOINT_ * (D_ + DEC_);
  __bf16* jw2b = bws; bws += (size_t)VOCAB_ * JOINT_;

  dim3 blk(256);
#define CVT(src, dst, n) cvtw_k<<<dim3(((n) + 255) / 256), blk, 0, stream>>>(src, dst, n)
#define GEMM(Ap, ldaV, Wp, ldwV, biasP, resP, outFp, outBp, ldoV, Mv, Nv, Kv, epiV, alphaV)  \
  gemm_wmma_bf16<<<dim3((Mv)/32, ((Nv)+255)/256), blk, 0, stream>>>(                         \
      Ap, ldaV, Wp, ldwV, biasP, resP, outFp, outBp, ldoV, Mv, Nv, Kv, epiV, alphaV)

  // ---- weight pre-conversion (once per launch; amortized over all M rows) ----
  CVT(lin_w, linb, D_ * D_);
  for (int l = 0; l < L_; ++l)
    for (int wq = 0; wq < 8; ++wq)
      CVT(P[4 + l * 30 + wIdx[wq]], Wb[l][wq], wSz[wq]);
  CVT(jw1, jw1b, JOINT_ * (D_ + DEC_));
  CVT(jw2, jw2b, VOCAB_ * JOINT_);

  // ---- encoder front ----
  subsample_k<<<dim3(BT), blk, 0, stream>>>(x, sub_w, sub_b, T1b);
  GEMM(T1b, D_, linb, D_, lin_b, nullptr, X, (__bf16*)nullptr, D_, BT, D_, D_, EPI_NONE, 0.f);

  // ---- conformer stack ----
  for (int l = 0; l < L_; ++l) {
    const float* const* Lp = &P[4 + l * 30];
    __bf16* const* Wl = Wb[l];
    // ffn1: x += 0.5*FFN(LN(x))
    layernorm_bf16_k<<<dim3(BT), blk, 0, stream>>>(X, Lp[0], Lp[1], T1b);
    GEMM(T1b, D_, Wl[0], D_, Lp[3], nullptr, (float*)nullptr, T2b, FFN_, BT, FFN_, D_, EPI_SILU, 0.f);
    GEMM(T2b, FFN_, Wl[1], FFN_, Lp[5], X, X, (__bf16*)nullptr, D_, BT, D_, FFN_, EPI_RES, 0.5f);
    // mha: x += Out(Attn(LN(x)))
    layernorm_bf16_k<<<dim3(BT), blk, 0, stream>>>(X, Lp[6], Lp[7], T1b);
    GEMM(T1b, D_, Wl[2], D_, Lp[9], nullptr, T2, (__bf16*)nullptr, 3 * D_, BT, 3 * D_, D_, EPI_NONE, 0.f);
    attention_k<<<dim3(T_, B_ * H_), blk, 0, stream>>>(T2, xlen, T3b);
    GEMM(T3b, D_, Wl[3], D_, Lp[11], X, X, (__bf16*)nullptr, D_, BT, D_, D_, EPI_RES, 1.f);
    // conv module: x += PW2(SiLU(BN(DW(GLU(PW1(LN(x)))))))
    layernorm_bf16_k<<<dim3(BT), blk, 0, stream>>>(X, Lp[12], Lp[13], T1b);
    GEMM(T1b, D_, Wl[4], D_, Lp[15], nullptr, T2, (__bf16*)nullptr, 2 * D_, BT, 2 * D_, D_, EPI_NONE, 0.f);
    glu_k<<<dim3((BT * D_ + 255) / 256), blk, 0, stream>>>(T2, T3);
    dwconv_k<<<dim3(BT), blk, 0, stream>>>(T3, Lp[16], Lp[17], Lp[18], Lp[19], T3b);
    GEMM(T3b, D_, Wl[5], D_, Lp[21], X, X, (__bf16*)nullptr, D_, BT, D_, D_, EPI_RES, 1.f);
    // ffn2
    layernorm_bf16_k<<<dim3(BT), blk, 0, stream>>>(X, Lp[22], Lp[23], T1b);
    GEMM(T1b, D_, Wl[6], D_, Lp[25], nullptr, (float*)nullptr, T2b, FFN_, BT, FFN_, D_, EPI_SILU, 0.f);
    GEMM(T2b, FFN_, Wl[7], FFN_, Lp[27], X, X, (__bf16*)nullptr, D_, BT, D_, FFN_, EPI_RES, 0.5f);
    // final LN (f32, feeds next layer's residual stream)
    layernorm_k<<<dim3(BT), blk, 0, stream>>>(X, Lp[28], Lp[29], X);
  }

  // ---- decoder ----
  embed_k<<<dim3((BU * DEC_ + 255) / 256), blk, 0, stream>>>(y, emb, EMBb);
  lstm_k<<<dim3(1), dim3(1024), 0, stream>>>(EMBb, w_ih, w_hh, b_ih, b_hh, DEC16);

  // ---- joint ----
  CVT(X, Xb, BT * D_);   // enc -> bf16 for e-projection
  // e' = enc @ w1e^T + jb1   (jb1 folded here so joint loop skips it)
  GEMM(Xb, D_, jw1b, D_ + DEC_, jb1, nullptr, Eb, (__bf16*)nullptr, JOINT_, BT, JOINT_, D_, EPI_NONE, 0.f);
  // d = dec @ w1d^T   (w1d = jw1[:, D:], row stride 512)
  GEMM(DEC16, DEC_, jw1b + D_, D_ + DEC_, nullptr, nullptr, Db, (__bf16*)nullptr, JOINT_, BU, JOINT_, DEC_, EPI_NONE, 0.f);
  // logits: relu(e'+d) @ jw2^T + jb2 -> [B,T,U,VOCAB]
  joint_wmma<<<dim3((B_ * T_ * U_) / 32, 2), blk, 0, stream>>>(
      Eb, Db, jw2b, jb2, (float*)d_out);
#undef GEMM
#undef CVT
}
